// Mamba_77446850282160
// MI455X (gfx1250) — compile-verified, matrix operand order checked by reading, not run on hardware
//
#include <hip/hip_runtime.h>
#include <hip/hip_bf16.h>
#include <math.h>

// ---------------- problem dimensions (compile-time) ----------------
#define D_MODEL  256
#define D_STATES 48
#define D_CONV   4
#define D_IN     512            // EXPAND * D_MODEL
#define DT_RANK  16             // D_MODEL / 16
#define BATCH    2
#define SEQ      2048
#define M_ROWS   (BATCH * SEQ)  // 4096
#define N_IN     (2 * D_IN)     // 1024
#define N_XDBL   (DT_RANK + 2 * D_STATES)  // 112

typedef __attribute__((ext_vector_type(16))) _Float16 v16h;
typedef __attribute__((ext_vector_type(8)))  float    v8f;

__device__ __forceinline__ float silu_f(float x)     { return x / (1.0f + __expf(-x)); }
__device__ __forceinline__ float softplus_f(float x) { return (x > 20.0f) ? x : log1pf(__expf(x)); }

// K-pair base inside a 16x32 (or 32x16) f16 WMMA fragment, per ISA 7.12.2:
// VGPR v holds halves {kb, kb+1}, kb = (v<4 ? 2v : 16+2(v-4)) + 8*(lane>>4)
__device__ __forceinline__ int frag_kb(int v, int half)
{
    return ((v < 4) ? (2 * v) : (16 + 2 * (v - 4))) + 8 * half;
}

// =====================================================================
// Packing kernels: f32 matrices -> f16 fragment-linear tiles.
// Layout: tile t, lane l -> 16 contiguous halves at P[(t*32 + l)*16].
//   A-pack tiles indexed t = mt*KT + kt   (A is M x K, row major)
//   B-pack tiles indexed t = nt*KT + kt   (B is K x N, row major)
// =====================================================================
__global__ void __launch_bounds__(256)
pack_a_f16(const float* __restrict__ A, _Float16* __restrict__ P,
           int Mtiles, int KT, int lda)
{
    const int gid = blockIdx.x * blockDim.x + threadIdx.x;
    if (gid >= Mtiles * KT * 32) return;
    const int lane = gid & 31;
    const int tile = gid >> 5;
    const int kt   = tile % KT;
    const int mt   = tile / KT;
    const int half = lane >> 4;
    const int m    = lane & 15;
    const float* row = A + (size_t)(mt * 16 + m) * lda + kt * 32;
    v16h t;
#pragma unroll
    for (int v = 0; v < 8; ++v) {
        const int kb = frag_kb(v, half);
        t[2 * v]     = (_Float16)row[kb];
        t[2 * v + 1] = (_Float16)row[kb + 1];
    }
    *(v16h*)(P + (size_t)gid * 16) = t;
}

// same as pack_a but element = Y * silu(RES)  (fused SiLU gate)
__global__ void __launch_bounds__(256)
pack_a_gated_f16(const float* __restrict__ Y, const float* __restrict__ RES,
                 _Float16* __restrict__ P, int Mtiles, int KT, int lda)
{
    const int gid = blockIdx.x * blockDim.x + threadIdx.x;
    if (gid >= Mtiles * KT * 32) return;
    const int lane = gid & 31;
    const int tile = gid >> 5;
    const int kt   = tile % KT;
    const int mt   = tile / KT;
    const int half = lane >> 4;
    const int m    = lane & 15;
    const size_t base = (size_t)(mt * 16 + m) * lda + kt * 32;
    v16h t;
#pragma unroll
    for (int v = 0; v < 8; ++v) {
        const int kb = frag_kb(v, half);
        t[2 * v]     = (_Float16)(Y[base + kb]     * silu_f(RES[base + kb]));
        t[2 * v + 1] = (_Float16)(Y[base + kb + 1] * silu_f(RES[base + kb + 1]));
    }
    *(v16h*)(P + (size_t)gid * 16) = t;
}

__global__ void __launch_bounds__(256)
pack_b_f16(const float* __restrict__ B, _Float16* __restrict__ P,
           int Ntiles, int KT, int ldb)
{
    const int gid = blockIdx.x * blockDim.x + threadIdx.x;
    if (gid >= Ntiles * KT * 32) return;
    const int lane = gid & 31;
    const int tile = gid >> 5;
    const int kt   = tile % KT;
    const int nt   = tile / KT;
    const int half = lane >> 4;
    const int n    = lane & 15;
    const float* col = B + (size_t)(kt * 32) * ldb + nt * 16 + n;
    v16h t;
#pragma unroll
    for (int v = 0; v < 8; ++v) {
        const int kb = frag_kb(v, half);
        t[2 * v]     = (_Float16)col[(size_t)kb * ldb];
        t[2 * v + 1] = (_Float16)col[(size_t)(kb + 1) * ldb];
    }
    *(v16h*)(P + (size_t)gid * 16) = t;
}

// C/D: VGPR r -> (M = r + 8*(lane>>4), N = lane&15)
__device__ __forceinline__ void wmma_store_c(float* __restrict__ C, int ldc,
                                             int row0, int col0, int lane, v8f c)
{
    const int half = lane >> 4;
    const int n    = lane & 15;
#pragma unroll
    for (int r = 0; r < 8; ++r)
        C[(size_t)(row0 + r + 8 * half) * ldc + (col0 + n)] = c[r];
}

// =====================================================================
// K1: x @ W_in -> xs | res   (M=4096, K=256 (KT=8), N=1024)
// Each wave computes a 16x64 strip: 4 accumulators share one A fragment.
// =====================================================================
__global__ void __launch_bounds__(256)
mamba_gemm_in(const _Float16* __restrict__ Ap, const _Float16* __restrict__ Bp,
              float* __restrict__ xs, float* __restrict__ res)
{
    __builtin_amdgcn_s_wait_tensorcnt(0);
    const int KT   = D_MODEL / 32;                   // 8
    const int lane = threadIdx.x;
    const int mt   = blockIdx.x;
    const int ng   = blockIdx.y * 8 + threadIdx.y;   // 0..15, 64-col group
    const int nt0  = ng * 4;

    const _Float16* ap = Ap + ((size_t)mt  * KT * 32 + lane) * 16;
    const _Float16* bp = Bp + ((size_t)nt0 * KT * 32 + lane) * 16;
    const size_t aStep = 32 * 16;          // halves per kt
    const size_t bStrd = (size_t)KT * 32 * 16;  // halves per n-tile

    v8f c0 = {}, c1 = {}, c2 = {}, c3 = {};
    for (int kt = 0; kt < KT; ++kt) {
        if (kt + 1 < KT) __builtin_prefetch(ap + aStep, 0, 0);
        v16h a  = *(const v16h*)ap;
        v16h b0 = *(const v16h*)(bp);
        v16h b1 = *(const v16h*)(bp + bStrd);
        v16h b2 = *(const v16h*)(bp + 2 * bStrd);
        v16h b3 = *(const v16h*)(bp + 3 * bStrd);
        c0 = __builtin_amdgcn_wmma_f32_16x16x32_f16(false, a, false, b0, (short)0, c0, false, false);
        c1 = __builtin_amdgcn_wmma_f32_16x16x32_f16(false, a, false, b1, (short)0, c1, false, false);
        c2 = __builtin_amdgcn_wmma_f32_16x16x32_f16(false, a, false, b2, (short)0, c2, false, false);
        c3 = __builtin_amdgcn_wmma_f32_16x16x32_f16(false, a, false, b3, (short)0, c3, false, false);
        ap += aStep;
        bp += aStep;
    }

    const int half = lane >> 4;
    const int n    = lane & 15;
    const int row0 = mt * 16;
    v8f acc[4] = {c0, c1, c2, c3};
#pragma unroll
    for (int t = 0; t < 4; ++t) {
        const int col = (nt0 + t) * 16 + n;
#pragma unroll
        for (int r = 0; r < 8; ++r) {
            const int row = row0 + r + 8 * half;
            const float v = acc[t][r];
            if (col < D_IN) xs [(size_t)row * D_IN + col]          = v;
            else            res[(size_t)row * D_IN + (col - D_IN)] = v;
        }
    }
}

// =====================================================================
// K3: xc @ W_x -> x_dbl   (M=4096, K=512 (KT=16), N=112: 7 tiles, 1/wave)
// =====================================================================
__global__ void __launch_bounds__(224)
mamba_gemm_xdbl(const _Float16* __restrict__ Ap, const _Float16* __restrict__ Bp,
                float* __restrict__ xdbl)
{
    const int KT   = D_IN / 32;                      // 16
    const int lane = threadIdx.x;
    const int mt   = blockIdx.x;
    const int nt   = threadIdx.y;                    // 0..6

    const _Float16* ap = Ap + ((size_t)mt * KT * 32 + lane) * 16;
    const _Float16* bp = Bp + ((size_t)nt * KT * 32 + lane) * 16;
    const size_t step = 32 * 16;

    v8f c = {};
    for (int kt = 0; kt < KT; ++kt) {
        if (kt + 1 < KT) __builtin_prefetch(ap + step, 0, 0);
        v16h a = *(const v16h*)ap;
        v16h b = *(const v16h*)bp;
        c = __builtin_amdgcn_wmma_f32_16x16x32_f16(false, a, false, b, (short)0, c, false, false);
        ap += step;
        bp += step;
    }
    wmma_store_c(xdbl, N_XDBL, mt * 16, nt * 16, lane, c);
}

// =====================================================================
// K6: (y * silu(res)) @ W_out -> out   (M=4096, K=512 (KT=16), N=256)
// =====================================================================
__global__ void __launch_bounds__(128)
mamba_gemm_out(const _Float16* __restrict__ Ap, const _Float16* __restrict__ Bp,
               float* __restrict__ out)
{
    const int KT   = D_IN / 32;                      // 16
    const int lane = threadIdx.x;
    const int mt   = blockIdx.x;
    const int nt0  = threadIdx.y * 4;                // 4 waves x 4 tiles = N=256

    const _Float16* ap = Ap + ((size_t)mt  * KT * 32 + lane) * 16;
    const _Float16* bp = Bp + ((size_t)nt0 * KT * 32 + lane) * 16;
    const size_t aStep = 32 * 16;
    const size_t bStrd = (size_t)KT * 32 * 16;

    v8f c0 = {}, c1 = {}, c2 = {}, c3 = {};
    for (int kt = 0; kt < KT; ++kt) {
        if (kt + 1 < KT) __builtin_prefetch(ap + aStep, 0, 0);
        v16h a  = *(const v16h*)ap;
        v16h b0 = *(const v16h*)(bp);
        v16h b1 = *(const v16h*)(bp + bStrd);
        v16h b2 = *(const v16h*)(bp + 2 * bStrd);
        v16h b3 = *(const v16h*)(bp + 3 * bStrd);
        c0 = __builtin_amdgcn_wmma_f32_16x16x32_f16(false, a, false, b0, (short)0, c0, false, false);
        c1 = __builtin_amdgcn_wmma_f32_16x16x32_f16(false, a, false, b1, (short)0, c1, false, false);
        c2 = __builtin_amdgcn_wmma_f32_16x16x32_f16(false, a, false, b2, (short)0, c2, false, false);
        c3 = __builtin_amdgcn_wmma_f32_16x16x32_f16(false, a, false, b3, (short)0, c3, false, false);
        ap += aStep;
        bp += aStep;
    }
    v8f acc[4] = {c0, c1, c2, c3};
#pragma unroll
    for (int t = 0; t < 4; ++t)
        wmma_store_c(out, D_MODEL, mt * 16, (nt0 + t) * 16, lane, acc[t]);
}

// ---------------- K2: depthwise causal conv(4) + bias + SiLU --------------
__global__ void __launch_bounds__(256)
mamba_conv_silu(const float* __restrict__ xs, const float* __restrict__ ck,
                const float* __restrict__ cb, float* __restrict__ xc)
{
    const int idx = blockIdx.x * blockDim.x + threadIdx.x;
    if (idx >= M_ROWS * D_IN) return;
    const int d = idx & (D_IN - 1);
    const int l = (idx / D_IN) & (SEQ - 1);
    float acc = cb[d];
#pragma unroll
    for (int j = 0; j < D_CONV; ++j) {
        const int ls = l - (D_CONV - 1) + j;
        if (ls >= 0) acc += xs[idx + (j - (D_CONV - 1)) * D_IN] * ck[j * D_IN + d];
    }
    xc[idx] = silu_f(acc);
}

// ---------------- K4: delta = softplus(x_dbl[:,:16] @ W_dt + b_dt) --------
__global__ void __launch_bounds__(256)
mamba_delta(const float* __restrict__ xdbl, const float* __restrict__ Wdt,
            const float* __restrict__ bdt, float* __restrict__ delta)
{
    const int idx = blockIdx.x * blockDim.x + threadIdx.x;
    if (idx >= M_ROWS * D_IN) return;
    const int row = idx / D_IN;
    const int d   = idx & (D_IN - 1);
    const float* xr = xdbl + (size_t)row * N_XDBL;
    float acc = bdt[d];
#pragma unroll
    for (int r = 0; r < DT_RANK; ++r) acc += xr[r] * Wdt[r * D_IN + d];
    delta[idx] = softplus_f(acc);
}

// ---------------- K5: selective scan, one wave32 per (b, d) ---------------
__global__ void __launch_bounds__(256)
mamba_scan(const float* __restrict__ xc, const float* __restrict__ delta,
           const float* __restrict__ xdbl, const float* __restrict__ A_log,
           const float* __restrict__ Dv, float* __restrict__ y)
{
    const int lane = threadIdx.x;
    const int wave = blockIdx.x * blockDim.y + threadIdx.y;   // 0..1023
    const int b    = wave / D_IN;
    const int d    = wave & (D_IN - 1);

    const int  n1   = lane;
    const int  n2   = lane + 32;
    const bool act2 = (n2 < D_STATES);                         // lanes 0..15

    const float A1 = -__expf(A_log[(size_t)d * D_STATES + n1]);
    const float A2 = act2 ? -__expf(A_log[(size_t)d * D_STATES + n2]) : 0.0f;
    const float Dd = Dv[d];

    float s1 = 0.0f, s2 = 0.0f;
    const size_t base = (size_t)b * SEQ * D_IN + d;
    const size_t rowb = (size_t)b * SEQ;

    for (int l = 0; l < SEQ; ++l) {
        const float dt = delta[base + (size_t)l * D_IN];
        const float u  = xc   [base + (size_t)l * D_IN];
        const float* xr = xdbl + (rowb + l) * N_XDBL;
        const float B1 = xr[DT_RANK + n1];
        const float C1 = xr[DT_RANK + D_STATES + n1];
        const float B2 = act2 ? xr[DT_RANK + n2] : 0.0f;
        const float C2 = act2 ? xr[DT_RANK + D_STATES + n2] : 0.0f;

        s1 = s1 * __expf(dt * A1) + dt * u * B1;
        s2 = s2 * __expf(dt * A2) + dt * u * B2;   // lanes>=16: stays 0

        float acc = s1 * C1 + s2 * C2;
#pragma unroll
        for (int m = 16; m > 0; m >>= 1) acc += __shfl_xor(acc, m, 32);
        if (lane == 0) y[base + (size_t)l * D_IN] = acc + u * Dd;
    }
}

// ---------------- host-side launch ---------------------------------------
extern "C" void kernel_launch(void* const* d_in, const int* in_sizes, int n_in,
                              void* d_out, int out_size, void* d_ws, size_t ws_size,
                              hipStream_t stream)
{
    const float* x      = (const float*)d_in[0];
    const float* W_in   = (const float*)d_in[1];
    const float* conv_k = (const float*)d_in[2];
    const float* conv_b = (const float*)d_in[3];
    const float* W_x    = (const float*)d_in[4];
    const float* W_dt   = (const float*)d_in[5];
    const float* b_dt   = (const float*)d_in[6];
    const float* A_log  = (const float*)d_in[7];
    const float* Dv     = (const float*)d_in[8];
    const float* W_out  = (const float*)d_in[9];
    float* out = (float*)d_out;

    // ---- workspace layout: f32 intermediates then packed f16 tiles ----
    float* ws    = (float*)d_ws;
    float* xs    = ws;
    float* res   = xs    + (size_t)M_ROWS * D_IN;
    float* xc    = res   + (size_t)M_ROWS * D_IN;
    float* xdbl  = xc    + (size_t)M_ROWS * D_IN;
    float* delta = xdbl  + (size_t)M_ROWS * N_XDBL;
    float* y     = delta + (size_t)M_ROWS * D_IN;
    _Float16* hbase = (_Float16*)(y + (size_t)M_ROWS * D_IN);
    _Float16* Xp    = hbase;                                  // 4096x256
    _Float16* WinP  = Xp   + (size_t)M_ROWS * D_MODEL;        // 256x1024
    _Float16* xcP   = WinP + (size_t)D_MODEL * N_IN;          // 4096x512
    _Float16* WxP   = xcP  + (size_t)M_ROWS * D_IN;           // 512x112
    _Float16* ygP   = WxP  + (size_t)D_IN * N_XDBL;           // 4096x512
    _Float16* WoutP = ygP  + (size_t)M_ROWS * D_IN;           // 512x256

    const int Mtiles = M_ROWS / 16;      // 256
    const int KT_in  = D_MODEL / 32;     // 8
    const int KT_din = D_IN / 32;        // 16

    // ---- pack weights + x ----
    pack_a_f16<<<(Mtiles * KT_in * 32 + 255) / 256, 256, 0, stream>>>(x, Xp, Mtiles, KT_in, D_MODEL);
    pack_b_f16<<<((N_IN / 16) * KT_in * 32 + 255) / 256, 256, 0, stream>>>(W_in, WinP, N_IN / 16, KT_in, N_IN);
    pack_b_f16<<<((N_XDBL / 16) * KT_din * 32 + 255) / 256, 256, 0, stream>>>(W_x, WxP, N_XDBL / 16, KT_din, N_XDBL);
    pack_b_f16<<<((D_MODEL / 16) * KT_din * 32 + 255) / 256, 256, 0, stream>>>(W_out, WoutP, D_MODEL / 16, KT_din, D_MODEL);

    // K1: x @ W_in -> xs,res   (grid covers 16 col-groups of 64)
    mamba_gemm_in<<<dim3(Mtiles, 2), dim3(32, 8), 0, stream>>>(Xp, WinP, xs, res);
    // K2: conv + SiLU
    mamba_conv_silu<<<(M_ROWS * D_IN + 255) / 256, 256, 0, stream>>>(xs, conv_k, conv_b, xc);
    // pack xc (A of next GEMM)
    pack_a_f16<<<(Mtiles * KT_din * 32 + 255) / 256, 256, 0, stream>>>(xc, xcP, Mtiles, KT_din, D_IN);
    // K3: xc @ W_x -> x_dbl
    mamba_gemm_xdbl<<<dim3(Mtiles), dim3(32, 7), 0, stream>>>(xcP, WxP, xdbl);
    // K4: delta
    mamba_delta<<<(M_ROWS * D_IN + 255) / 256, 256, 0, stream>>>(xdbl, W_dt, b_dt, delta);
    // K5: selective scan (1024 wave32 scans, 8 waves/block)
    mamba_scan<<<dim3(BATCH * D_IN / 8), dim3(32, 8), 0, stream>>>(xc, delta, xdbl, A_log, Dv, y);
    // pack gated y
    pack_a_gated_f16<<<(Mtiles * KT_din * 32 + 255) / 256, 256, 0, stream>>>(y, res, ygP, Mtiles, KT_din, D_IN);
    // K6: gated output GEMM
    mamba_gemm_out<<<dim3(Mtiles), dim3(32, 4), 0, stream>>>(ygP, WoutP, out);
}
